// MultiHeadAttention_67972152427020
// MI455X (gfx1250) — matrix operand order repaired and checked by reading, not compiled, not run on hardware
//
#include <hip/hip_runtime.h>

// ---------------------------------------------------------------------------
// MHA forward for MI455X (gfx1250): bf16 WMMA everywhere, fp32 accumulate.
// Problem: B=2, S=2048, D=1024, H=16, Dh=64.  Compute-bound (~70 GFLOP vs
// ~50MB of traffic -> memory time ~2us at 23.3TB/s), so all matmuls go
// through v_wmma_f32_16x16x32_bf16.  Projection GEMMs stage the shared W
// tile in LDS (8x reduction of B-side global traffic per block).
// ---------------------------------------------------------------------------

typedef __attribute__((ext_vector_type(16))) __bf16 v16bf;
typedef __attribute__((ext_vector_type(8)))  float  v8f;

#define DEVI __device__ __forceinline__

constexpr int Bsz  = 2;
constexpr int Seq  = 2048;
constexpr int Dm   = 1024;
constexpr int H    = 16;
constexpr int Dh   = 64;
constexpr int Mtok = Bsz * Seq;   // 4096 tokens

DEVI unsigned short f2bf(float f) {
    unsigned int u = __float_as_uint(f);
    u += 0x7FFFu + ((u >> 16) & 1u);          // round-to-nearest-even
    return (unsigned short)(u >> 16);
}

union FragBF {                                // 32 bytes = 16 bf16 = one WMMA operand
    uint4 u4[2];
    v16bf v;
};

DEVI v8f wmma_bf16(const v16bf& a, const v16bf& b, v8f c) {
    return __builtin_amdgcn_wmma_f32_16x16x32_bf16(
        /*neg_a=*/false, a, /*neg_b=*/false, b,
        /*c_mod=*/(short)0, c, /*reuse_a=*/false, /*reuse_b=*/false);
}

// ---------------------------------------------------------------------------
// fp32 -> bf16 conversion (vectorized; n divisible by 4)
// ---------------------------------------------------------------------------
__global__ void f32_to_bf16(const float* __restrict__ in,
                            unsigned short* __restrict__ out, int n) {
    int i = (blockIdx.x * blockDim.x + threadIdx.x) * 4;
    if (i + 4 <= n) {
        float4 v = *(const float4*)(in + i);
        uint2 u;
        u.x = (unsigned)f2bf(v.x) | ((unsigned)f2bf(v.y) << 16);
        u.y = (unsigned)f2bf(v.z) | ((unsigned)f2bf(v.w) << 16);
        *(uint2*)(out + i) = u;
    }
}

// ---------------------------------------------------------------------------
// GEMM: C[M,N] = A[M,K](bf16) * W[N,K](bf16)^T   (the x @ w.T pattern)
// Block tile: M=128 (8 waves x 16 rows) x N=64.  All waves share the same
// 64x32 B tile per k-step -> stage it once in LDS (4KB, 1 b128/thread) with
// a register prefetch of the next tile; waves read fragments via ds_load_b128.
// LDS row stride = 40 halves (80B): 16B-aligned and the 16 lanes of each
// half land on 16 distinct banks (row*20 dwords mod 64 are all distinct).
// Fragment layouts per CDNA5 ISA 7.12.2:
//   A  lane L: row M=L&15, halves at K = kb..kb+7 and 16+kb..16+kb+7, kb=(L<16?0:8)
//   B  lane L: col N=L&15, halves at K = kb..kb+7 and 16+kb..16+kb+7, kb=(L<16?0:8)
//   C  lane L: col N=L&15, rows M = r + (L<16?0:8) for vgpr r
// MODE 0: store bf16 row-major | MODE 1: store bf16 transposed (V^T)
// MODE 2: store fp32 row-major (final output)
// ---------------------------------------------------------------------------
template <int MODE>
__global__ void __launch_bounds__(256)
gemm_bf16_wmma(const unsigned short* __restrict__ A,
               const unsigned short* __restrict__ W,
               void* __restrict__ outp,
               int M, int N, int K, int ldo) {
    constexpr int BSTRIDE = 40;                        // halves per LDS row (80B)
    __shared__ __align__(16) unsigned short Bs[64 * BSTRIDE];   // 5KB

    const int tid  = threadIdx.x;
    const int lane = tid & 31;
    const int wv   = tid >> 5;
    const int lh   = lane & 15;
    const int khA  = (lane < 16) ? 0 : 8;
    const int khB  = (lane < 16) ? 0 : 16;
    const int hb   = (lane < 16) ? 0 : 8;

    const int m0 = (blockIdx.y * 8 + wv) * 16;         // grid tiles M exactly
    const int n0 = blockIdx.x * 64;

    // cooperative B loader: thread -> (row 0..63, 16B chunk 0..3) of the tile
    const int brow = tid >> 2;
    const int bchk = (tid & 3) * 8;
    const unsigned short* wptr = W + (size_t)(n0 + brow) * K + bchk;

    v8f acc[4];
#pragma unroll
    for (int t = 0; t < 4; ++t) acc[t] = v8f{0.f,0.f,0.f,0.f,0.f,0.f,0.f,0.f};

    const unsigned short* arow = A + (size_t)(m0 + lh) * K;

    uint4 breg = *(const uint4*)(wptr);                // prefetch k-step 0

    for (int k0 = 0; k0 < K; k0 += 32) {
        __syncthreads();                               // prior reads of Bs done
        *(uint4*)(&Bs[brow * BSTRIDE + bchk]) = breg;
        if (k0 + 32 < K)                               // prefetch next tile
            breg = *(const uint4*)(wptr + k0 + 32);
        FragBF a;
        a.u4[0] = *(const uint4*)(arow + k0 + khA);
        a.u4[1] = *(const uint4*)(arow + k0 + 16 + khA);
        __syncthreads();                               // Bs stores visible
#pragma unroll
        for (int t = 0; t < 4; ++t) {
            FragBF b;
            const unsigned short* bs = &Bs[(t * 16 + lh) * BSTRIDE + khA];
            b.u4[0] = *(const uint4*)(bs);
            b.u4[1] = *(const uint4*)(bs + 16);
            acc[t] = wmma_bf16(a.v, b.v, acc[t]);
        }
    }

#pragma unroll
    for (int t = 0; t < 4; ++t) {
#pragma unroll
        for (int r = 0; r < 8; ++r) {
            const int row = m0 + r + hb;
            const int col = n0 + t * 16 + lh;
            if (MODE == 2) {
                ((float*)outp)[(size_t)row * ldo + col] = acc[t][r];
            } else if (MODE == 0) {
                ((unsigned short*)outp)[(size_t)row * ldo + col] = f2bf(acc[t][r]);
            } else {
                ((unsigned short*)outp)[(size_t)col * ldo + row] = f2bf(acc[t][r]);
            }
        }
    }
}

// ---------------------------------------------------------------------------
// Causal flash attention. One wave owns a 16-query tile of one (b,h).
//   Q,K : bf16 [B*S, 1024] row-major (head h occupies cols h*64..h*64+63)
//   Vt  : bf16 [1024, B*S] (model-dim major -> P*V B-frags load contiguously)
//   O   : bf16 [B*S, 1024]
// Online softmax over 32-key steps; P goes C-layout -> LDS -> A-layout.
// ---------------------------------------------------------------------------
__global__ void __launch_bounds__(256)
attn_causal(const unsigned short* __restrict__ Q,
            const unsigned short* __restrict__ Km,
            const unsigned short* __restrict__ Vt,
            unsigned short* __restrict__ O) {
    __shared__ __align__(16) unsigned short P[8][16 * 32];   // per-wave P tile

    const int lane = threadIdx.x & 31;
    const int wv   = threadIdx.x >> 5;
    const int lh   = lane & 15;
    const int khA  = (lane < 16) ? 0 : 8;
    const int khB  = (lane < 16) ? 0 : 16;
    const int hb   = (lane < 16) ? 0 : 8;

    const int b       = blockIdx.z;
    const int h       = blockIdx.y;
    const int qt      = blockIdx.x * 8 + wv;
    const int qbase   = qt * 16;
    const int rowbase = b * Seq;
    const int col0    = h * Dh;

    // Q fragments for d 0..31 and 32..63 (A-layout, loaded once)
    FragBF aq0, aq1;
    {
        const unsigned short* qp = Q + (size_t)(rowbase + qbase + lh) * Dm + col0;
        aq0.u4[0] = *(const uint4*)(qp + khA);
        aq0.u4[1] = *(const uint4*)(qp + 16 + khA);
        aq1.u4[0] = *(const uint4*)(qp + 32 + khA);
        aq1.u4[1] = *(const uint4*)(qp + 48 + khA);
    }

    v8f o[4];
#pragma unroll
    for (int t = 0; t < 4; ++t) o[t] = v8f{0.f,0.f,0.f,0.f,0.f,0.f,0.f,0.f};
    float mrow[8], lrow[8];
#pragma unroll
    for (int r = 0; r < 8; ++r) { mrow[r] = -1e30f; lrow[r] = 0.f; }

    for (int j = 0; j < qbase + 16; j += 32) {       // 32 keys per step
        float s0a[8], s1a[8];
#pragma unroll
        for (int g = 0; g < 2; ++g) {                // two 16-key groups
            const int kb = j + g * 16;
            const unsigned short* kp =
                Km + (size_t)(rowbase + kb + lh) * Dm + col0 + khA;
            FragBF bk0, bk1;
            bk0.u4[0] = *(const uint4*)(kp);
            bk0.u4[1] = *(const uint4*)(kp + 16);
            bk1.u4[0] = *(const uint4*)(kp + 32);
            bk1.u4[1] = *(const uint4*)(kp + 48);
            v8f s = v8f{0.f,0.f,0.f,0.f,0.f,0.f,0.f,0.f};
            s = wmma_bf16(aq0.v, bk0.v, s);          // d = 0..31
            s = wmma_bf16(aq1.v, bk1.v, s);          // d = 32..63
#pragma unroll
            for (int r = 0; r < 8; ++r) {
                const int qrow = qbase + r + hb;
                const int kcol = kb + lh;
                float val = s[r] * 0.125f;           // 1/sqrt(64)
                if (kcol > qrow) val = -1e30f;       // causal mask
                if (g) s1a[r] = val; else s0a[r] = val;
            }
        }
        // online softmax per row (row lives across the 16 lanes of a half)
#pragma unroll
        for (int r = 0; r < 8; ++r) {
            float mx = fmaxf(s0a[r], s1a[r]);
            mx = fmaxf(mx, __shfl_xor(mx, 1));
            mx = fmaxf(mx, __shfl_xor(mx, 2));
            mx = fmaxf(mx, __shfl_xor(mx, 4));
            mx = fmaxf(mx, __shfl_xor(mx, 8));
            const float nm = fmaxf(mrow[r], mx);
            const float al = __expf(mrow[r] - nm);
            const float p0 = __expf(s0a[r] - nm);
            const float p1 = __expf(s1a[r] - nm);
            float rs = p0 + p1;
            rs += __shfl_xor(rs, 1);
            rs += __shfl_xor(rs, 2);
            rs += __shfl_xor(rs, 4);
            rs += __shfl_xor(rs, 8);
            lrow[r] = lrow[r] * al + rs;
            mrow[r] = nm;
            const int rl = r + hb;
            P[wv][rl * 32 + lh]      = f2bf(p0);     // C-layout -> LDS
            P[wv][rl * 32 + 16 + lh] = f2bf(p1);
#pragma unroll
            for (int t = 0; t < 4; ++t) o[t][r] *= al;
        }
        // re-load P in A-layout (LDS ops are in-order within a wave)
        FragBF ap;
        ap.u4[0] = *(const uint4*)(&P[wv][lh * 32 + khA]);
        ap.u4[1] = *(const uint4*)(&P[wv][lh * 32 + 16 + khA]);
        // O += P * V  (V^T rows are model-dims -> contiguous key loads)
#pragma unroll
        for (int t = 0; t < 4; ++t) {
            const unsigned short* vp =
                Vt + (size_t)(col0 + t * 16 + lh) * Mtok + rowbase + j + khA;
            FragBF bv;
            bv.u4[0] = *(const uint4*)(vp);
            bv.u4[1] = *(const uint4*)(vp + 16);
            o[t] = wmma_bf16(ap.v, bv.v, o[t]);
        }
    }

    unsigned short* op = O + (size_t)(rowbase + qbase) * Dm + col0;
#pragma unroll
    for (int t = 0; t < 4; ++t) {
#pragma unroll
        for (int r = 0; r < 8; ++r) {
            const int rl = r + hb;
            op[(size_t)rl * Dm + t * 16 + lh] = f2bf(o[t][r] / lrow[r]);
        }
    }
}

// ---------------------------------------------------------------------------
// Host-side launch. Workspace layout (bf16 elements):
//   xh[4M] wqh[1M] wkh[1M] wvh[1M] woh[1M] qh[4M] kh[4M] vth[4M] oh[4M]
//   = 24M bf16 = 48 MB
// ---------------------------------------------------------------------------
extern "C" void kernel_launch(void* const* d_in, const int* in_sizes, int n_in,
                              void* d_out, int out_size, void* d_ws, size_t ws_size,
                              hipStream_t stream) {
    const float* x  = (const float*)d_in[0];
    const float* wq = (const float*)d_in[1];
    const float* wk = (const float*)d_in[2];
    const float* wv = (const float*)d_in[3];
    const float* wo = (const float*)d_in[4];

    unsigned short* ws = (unsigned short*)d_ws;
    size_t off = 0;
    unsigned short* xh  = ws + off; off += (size_t)Mtok * Dm;
    unsigned short* wqh = ws + off; off += (size_t)Dm * Dm;
    unsigned short* wkh = ws + off; off += (size_t)Dm * Dm;
    unsigned short* wvh = ws + off; off += (size_t)Dm * Dm;
    unsigned short* woh = ws + off; off += (size_t)Dm * Dm;
    unsigned short* qh  = ws + off; off += (size_t)Mtok * Dm;
    unsigned short* kh  = ws + off; off += (size_t)Mtok * Dm;
    unsigned short* vth = ws + off; off += (size_t)Mtok * Dm;
    unsigned short* oh  = ws + off; off += (size_t)Mtok * Dm;

    const int nx = Mtok * Dm;      // 4M
    const int nw = Dm * Dm;        // 1M
    f32_to_bf16<<<nx / 1024, 256, 0, stream>>>(x,  xh,  nx);
    f32_to_bf16<<<nw / 1024, 256, 0, stream>>>(wq, wqh, nw);
    f32_to_bf16<<<nw / 1024, 256, 0, stream>>>(wk, wkh, nw);
    f32_to_bf16<<<nw / 1024, 256, 0, stream>>>(wv, wvh, nw);
    f32_to_bf16<<<nw / 1024, 256, 0, stream>>>(wo, woh, nw);

    dim3 gblk(Dm / 64, Mtok / 128);   // (16, 32), 256 threads = 8 waves/block
    gemm_bf16_wmma<0><<<gblk, 256, 0, stream>>>(xh, wqh, qh,  Mtok, Dm, Dm, Dm);
    gemm_bf16_wmma<0><<<gblk, 256, 0, stream>>>(xh, wkh, kh,  Mtok, Dm, Dm, Dm);
    gemm_bf16_wmma<1><<<gblk, 256, 0, stream>>>(xh, wvh, vth, Mtok, Dm, Dm, Mtok);

    attn_causal<<<dim3(Seq / 16 / 8, H, Bsz), 256, 0, stream>>>(qh, kh, vth, oh);

    gemm_bf16_wmma<2><<<gblk, 256, 0, stream>>>(oh, woh, d_out, Mtok, Dm, Dm, Dm);
}
